// GroupQueryAttention_2319282340594
// MI455X (gfx1250) — matrix operand order repaired and checked by reading, not compiled, not run on hardware
//
#include <hip/hip_runtime.h>
#include <hip/hip_bf16.h>

// GQA attention on MI455X (gfx1250): f16 WMMA path, flash-style attention.
// HIDDEN=2048, H=32, HKV=8, D=64, B=2, S=2048, M=B*S=4096.

typedef __attribute__((ext_vector_type(16))) _Float16 v16h;
typedef __attribute__((ext_vector_type(8)))  _Float16 v8h;
typedef __attribute__((ext_vector_type(8)))  float    v8f;

#define HIDDEN 2048
#define SEQ    2048
#define BATCH  2
#define MROWS  (BATCH * SEQ)   // 4096
#define HEADS  32
#define KVHEADS 8
#define HDIM   64
#define KVDIM  (KVHEADS * HDIM)  // 512

__device__ __forceinline__ v8f wmma_f16(v16h a, v16h b, v8f c) {
    return __builtin_amdgcn_wmma_f32_16x16x32_f16(
        /*neg_a=*/false, a, /*neg_b=*/false, b,
        /*c_mod=*/(short)0, c, /*reuse_a=*/false, /*reuse_b=*/false);
}

// A operand (16x32 f16, M x K): lanes 0-15 row=m0+lane, K=k0..k0+15;
// lanes 16-31 same rows, K=k0+16..k0+31.  Two b128 loads.
__device__ __forceinline__ v16h load_A(const _Float16* __restrict__ base, int ld,
                                       int m0, int k0, int lane) {
    const int row = m0 + (lane & 15);
    const int kk  = k0 + ((lane >> 4) << 4);
    const _Float16* p = base + row * ld + kk;
    v16h a;
    ((v8h*)&a)[0] = *(const v8h*)(p);
    ((v8h*)&a)[1] = *(const v8h*)(p + 8);
    return a;
}

// B operand (32x16 f16, K x N) read from a [N][K] (row=n, contiguous K) array:
// lanes 0-15 col=n0+lane, K=k0..k0+15; lanes 16-31 same cols, K=k0+16..k0+31.
__device__ __forceinline__ v16h load_B(const _Float16* __restrict__ baseT, int ld,
                                       int n0, int k0, int lane) {
    const int col = n0 + (lane & 15);
    const int kk  = k0 + ((lane >> 4) << 4);
    const _Float16* p = baseT + col * ld + kk;
    v16h b;
    ((v8h*)&b)[0] = *(const v8h*)(p);
    ((v8h*)&b)[1] = *(const v8h*)(p + 8);
    return b;
}

// ---------------- prep kernels ----------------
__global__ void k_cvt_f16(const float* __restrict__ in, _Float16* __restrict__ out, int n) {
    int i = blockIdx.x * blockDim.x + threadIdx.x;
    if (i < n) out[i] = (_Float16)in[i];
}

// W [K][N] f32 -> Wt [N][K] f16
__global__ void k_transpose_f16(const float* __restrict__ in, _Float16* __restrict__ out,
                                int K, int N) {
    int i = blockIdx.x * blockDim.x + threadIdx.x;
    if (i < K * N) {
        int k = i / N, n = i - k * N;
        out[n * K + k] = (_Float16)in[i];
    }
}

// ---------------- WMMA GEMM: C[M,N] = A[M,K] @ W[K,N] + bias ----------------
// One wave computes a 32x64 tile (two row-groups share every B load).
// MODE 0: store f16 row-major (ldO).   MODE 1: store f16 as Vt[(b*512+n)][s].
// MODE 2: store f32 row-major (ldO).
template <int MODE>
__global__ void k_gemm(const _Float16* __restrict__ A, const _Float16* __restrict__ Wt,
                       const float* __restrict__ bias, void* __restrict__ Out,
                       int M, int N, int K, int ldO) {
    const int lane   = threadIdx.x;           // 32 threads = 1 wave
    const int ntiles = N >> 6;
    const int mt = blockIdx.x / ntiles;
    const int nt = blockIdx.x - mt * ntiles;
    const int m0 = mt << 5, n0 = nt << 6;

    v8f acc[2][4] = {};
    for (int k0 = 0; k0 < K; k0 += 32) {
        v16h aL = load_A(A, K, m0,      k0, lane);
        v16h aH = load_A(A, K, m0 + 16, k0, lane);
#pragma unroll
        for (int nc = 0; nc < 4; ++nc) {
            v16h b = load_B(Wt, K, n0 + nc * 16, k0, lane);
            acc[0][nc] = wmma_f16(aL, b, acc[0][nc]);
            acc[1][nc] = wmma_f16(aH, b, acc[1][nc]);
        }
    }
    const int laneHi = lane >> 4, nlo = lane & 15;
#pragma unroll
    for (int mh = 0; mh < 2; ++mh) {
#pragma unroll
        for (int nc = 0; nc < 4; ++nc) {
            const int col = n0 + nc * 16 + nlo;
            const float bv = bias[col];
#pragma unroll
            for (int j = 0; j < 8; ++j) {
                const int row = m0 + mh * 16 + j + 8 * laneHi;  // C/D layout: VGPR j -> M=j(+8)
                const float v = acc[mh][nc][j] + bv;
                if (MODE == 0) {
                    ((_Float16*)Out)[row * ldO + col] = (_Float16)v;
                } else if (MODE == 1) {
                    const int b = row >> 11, s = row & (SEQ - 1);
                    ((_Float16*)Out)[(b * KVDIM + col) * SEQ + s] = (_Float16)v;
                } else {
                    ((float*)Out)[row * ldO + col] = v;
                }
            }
        }
    }
}

// ---------------- flash GQA: one wave per (b,h,16-query tile) ----------------
// Everything computed TRANSPOSED so each softmax row (one query) lives in lanes
// q and q+16 only:  Ct = K·Qt  (A=K rows, B=Q rows),  Ot = Vt·Pt (A=V rows).
// Qh: [4096][2048] f16.  Kh: [4096][512] f16 (rows b*S+s).  Vt: [(b*8+hkv)*64+d][S] f16.
// AO: [4096][2048] f16.
__global__ void k_flash_gqa(const _Float16* __restrict__ Qh, const _Float16* __restrict__ Kh,
                            const _Float16* __restrict__ Vt, _Float16* __restrict__ AO) {
    const int lane = threadIdx.x;
    const int bid  = blockIdx.x;            // B * H * (S/16) = 8192
    const int qt = bid & 127;
    const int h  = (bid >> 7) & 31;
    const int b  = bid >> 12;
    const int hkv = h >> 2;
    const int qrow = b * SEQ + qt * 16;
    const int laneHi = lane >> 4, nlo = lane & 15;

    // Q as B operands (columns = queries), resident for the whole loop.
    const v16h bQ0 = load_B(Qh, HIDDEN, qrow, h * HDIM + 0,  lane);
    const v16h bQ1 = load_B(Qh, HIDDEN, qrow, h * HDIM + 32, lane);

    const _Float16* Kbh = Kh + (size_t)(b * SEQ) * KVDIM + hkv * HDIM;   // key rows, ld=512
    const _Float16* Vbh = Vt + (size_t)(b * KVDIM + hkv * HDIM) * SEQ;   // d rows, ld=2048

    float m = -__builtin_inff();
    float l = 0.f;
    v8f acc[4] = {};   // Ot: rows d = nc*16 + j + 8*laneHi, col q = nlo
    const float scale = 0.125f;   // 1/sqrt(64)

    for (int j0 = 0; j0 < SEQ; j0 += 32) {
        // transposed scores: c0 rows = keys j0+j+8*laneHi, c1 rows = +16; col = query nlo
        v8f c0 = {}, c1 = {};
        c0 = wmma_f16(load_A(Kbh, KVDIM, j0,      0,  lane), bQ0, c0);
        c0 = wmma_f16(load_A(Kbh, KVDIM, j0,      32, lane), bQ1, c0);
        c1 = wmma_f16(load_A(Kbh, KVDIM, j0 + 16, 0,  lane), bQ0, c1);
        c1 = wmma_f16(load_A(Kbh, KVDIM, j0 + 16, 32, lane), bQ1, c1);

        float x0[8], x1[8];
        float mx = -__builtin_inff();
#pragma unroll
        for (int j = 0; j < 8; ++j) {
            x0[j] = c0[j] * scale;
            x1[j] = c1[j] * scale;
            mx = fmaxf(mx, fmaxf(x0[j], x1[j]));
        }
        mx = fmaxf(mx, __shfl_xor(mx, 16, 32));          // combine the two half-rows
        const float mnew  = fmaxf(m, mx);
        const float alpha = __expf(m - mnew);

        float p0[8], p1[8], ps = 0.f;
#pragma unroll
        for (int j = 0; j < 8; ++j) {
            p0[j] = __expf(x0[j] - mnew);
            p1[j] = __expf(x1[j] - mnew);
            ps += p0[j] + p1[j];
        }
        ps += __shfl_xor(ps, 16, 32);
        l = l * alpha + ps;
        m = mnew;
#pragma unroll
        for (int nc = 0; nc < 4; ++nc) acc[nc] *= alpha;   // per-query scale: lane-local

        // Build Pt B operand (keys x queries): element e = key j0 + laneHi*16 + e.
        // lo lane owns keys j0+0..7 (p0), needs partner's p0 (keys +8..15);
        // hi lane owns keys j0+24..31 (p1), needs partner's p1 (keys +16..23).
        unsigned own0[4], own1[4], recv[4];
        union { unsigned u; _Float16 f[2]; } pk;
#pragma unroll
        for (int i = 0; i < 4; ++i) {
            pk.f[0] = (_Float16)p0[2 * i]; pk.f[1] = (_Float16)p0[2 * i + 1]; own0[i] = pk.u;
            pk.f[0] = (_Float16)p1[2 * i]; pk.f[1] = (_Float16)p1[2 * i + 1]; own1[i] = pk.u;
        }
#pragma unroll
        for (int i = 0; i < 4; ++i) {
            unsigned send = laneHi ? own0[i] : own1[i];    // send what the partner needs
            recv[i] = __shfl_xor(send, 16, 32);
        }
        union { v16h h; unsigned u[8]; } bp;
#pragma unroll
        for (int i = 0; i < 4; ++i) {
            bp.u[i]     = laneHi ? recv[i] : own0[i];
            bp.u[4 + i] = laneHi ? own1[i] : recv[i];
        }

        // Ot += Vt(16d x 32keys) · Pt(32keys x 16q), per 16-d chunk
#pragma unroll
        for (int nc = 0; nc < 4; ++nc)
            acc[nc] = wmma_f16(load_A(Vbh, SEQ, nc * 16, j0, lane), bp.h, acc[nc]);
    }

    const float inv = 1.0f / l;
    const size_t orow = (size_t)(qrow + nlo) * HIDDEN + h * HDIM;
#pragma unroll
    for (int nc = 0; nc < 4; ++nc)
#pragma unroll
        for (int j = 0; j < 8; ++j)
            AO[orow + nc * 16 + j + 8 * laneHi] = (_Float16)(acc[nc][j] * inv);
}

// ---------------- launch ----------------
extern "C" void kernel_launch(void* const* d_in, const int* in_sizes, int n_in,
                              void* d_out, int out_size, void* d_ws, size_t ws_size,
                              hipStream_t stream) {
    const float* X  = (const float*)d_in[0];
    const float* Wq = (const float*)d_in[1];
    const float* bq = (const float*)d_in[2];
    const float* Wk = (const float*)d_in[3];
    const float* bk = (const float*)d_in[4];
    const float* Wv = (const float*)d_in[5];
    const float* bv = (const float*)d_in[6];
    const float* Wo = (const float*)d_in[7];
    const float* bo = (const float*)d_in[8];
    float* out = (float*)d_out;

    // workspace layout (f16, 256B-aligned slabs), total ~76 MB
    char* ws = (char*)d_ws;
    size_t off = 0;
    auto alloc = [&](size_t bytes) { char* p = ws + off; off = (off + bytes + 255) & ~(size_t)255; return p; };
    _Float16* Xh  = (_Float16*)alloc((size_t)MROWS * HIDDEN * 2);
    _Float16* WqT = (_Float16*)alloc((size_t)HIDDEN * HIDDEN * 2);
    _Float16* WkT = (_Float16*)alloc((size_t)KVDIM * HIDDEN * 2);
    _Float16* WvT = (_Float16*)alloc((size_t)KVDIM * HIDDEN * 2);
    _Float16* WoT = (_Float16*)alloc((size_t)HIDDEN * HIDDEN * 2);
    _Float16* Qh  = (_Float16*)alloc((size_t)MROWS * HIDDEN * 2);
    _Float16* Kh  = (_Float16*)alloc((size_t)MROWS * KVDIM * 2);
    _Float16* Vt  = (_Float16*)alloc((size_t)MROWS * KVDIM * 2);
    _Float16* AO  = (_Float16*)alloc((size_t)MROWS * HIDDEN * 2);

    // prep: convert X, transpose+convert weights
    {
        int n = MROWS * HIDDEN;
        k_cvt_f16<<<(n + 255) / 256, 256, 0, stream>>>(X, Xh, n);
        k_transpose_f16<<<(HIDDEN * HIDDEN + 255) / 256, 256, 0, stream>>>(Wq, WqT, HIDDEN, HIDDEN);
        k_transpose_f16<<<(HIDDEN * KVDIM + 255) / 256, 256, 0, stream>>>(Wk, WkT, HIDDEN, KVDIM);
        k_transpose_f16<<<(HIDDEN * KVDIM + 255) / 256, 256, 0, stream>>>(Wv, WvT, HIDDEN, KVDIM);
        k_transpose_f16<<<(HIDDEN * HIDDEN + 255) / 256, 256, 0, stream>>>(Wo, WoT, HIDDEN, HIDDEN);
    }
    // projections (32x64 tiles per wave)
    k_gemm<0><<<(MROWS / 32) * (HIDDEN / 64), 32, 0, stream>>>(Xh, WqT, bq, Qh, MROWS, HIDDEN, HIDDEN, HIDDEN);
    k_gemm<0><<<(MROWS / 32) * (KVDIM / 64),  32, 0, stream>>>(Xh, WkT, bk, Kh, MROWS, KVDIM, HIDDEN, KVDIM);
    k_gemm<1><<<(MROWS / 32) * (KVDIM / 64),  32, 0, stream>>>(Xh, WvT, bv, Vt, MROWS, KVDIM, HIDDEN, 0);
    // attention
    k_flash_gqa<<<BATCH * HEADS * (SEQ / 16), 32, 0, stream>>>(Qh, Kh, Vt, AO);
    // output projection (f32 out)
    k_gemm<2><<<(MROWS / 32) * (HIDDEN / 64), 32, 0, stream>>>(AO, WoT, bo, out, MROWS, HIDDEN, HIDDEN, HIDDEN);
}
